// ESPACE_23871428232084
// MI455X (gfx1250) — matrix operand order repaired and checked by reading, not compiled, not run on hardware
//
#include <hip/hip_runtime.h>
#include <hip/hip_bf16.h>
#include <cmath>

// ---------------------------------------------------------------------------
// Shapes
// ---------------------------------------------------------------------------
#define BB   32
#define TT   21
#define NOBJ 100
#define NUP  30
#define FSC  540
#define FUP  537
#define NCTX 33
#define DECL 10
#define NBT  (BB*TT)          // 672
#define ROWS_UP (NBT*NUP)     // 20160
#define ROWS_SC (NBT*NOBJ)    // 67200
#define SEQT (TT-1)           // 20

#define PAD16(x) (((x)+15)&~15)
#define PAD32(x) (((x)+31)&~31)

typedef __attribute__((ext_vector_type(16))) _Float16 v16h;
typedef __attribute__((ext_vector_type(8)))  _Float16 v8h;
typedef __attribute__((ext_vector_type(8)))  float    v8f;

union V16U { v16h v; v8h h[2]; };

// ---------------------------------------------------------------------------
// f32 -> f16 converter with column slice + K zero-padding (A-matrix prep)
// ---------------------------------------------------------------------------
__global__ void k_cvt(const float* __restrict__ src, _Float16* __restrict__ dst,
                      long rows, int src_ld, int src_off, int ksrc, int kpad) {
  long idx = (long)blockIdx.x * blockDim.x + threadIdx.x;
  long total = rows * (long)kpad;
  if (idx >= total) return;
  long r = idx / kpad; int c = (int)(idx - r * kpad);
  float x = (c < ksrc) ? src[r * (long)src_ld + src_off + c] : 0.0f;
  dst[idx] = (_Float16)x;
}

// dec input slice: seq_input[:, 10:20, :] -> f16 [320 x 64]
__global__ void k_cvt_dec(const float* __restrict__ src, _Float16* __restrict__ dst) {
  int idx = blockIdx.x * blockDim.x + threadIdx.x;
  if (idx >= BB * DECL * 64) return;
  int r = idx >> 6, c = idx & 63;
  int b = r / DECL, t = r % DECL;
  dst[idx] = (_Float16)src[((size_t)b * SEQT + DECL + t) * 64 + c];
}

// weight [K x N] f32 -> transposed padded f16 [Npad x Kpad]
__global__ void k_wT(const float* __restrict__ w, _Float16* __restrict__ dst,
                     int K, int N, int Kpad, int Npad) {
  int idx = blockIdx.x * blockDim.x + threadIdx.x;
  int total = Npad * Kpad;
  if (idx >= total) return;
  int n = idx / Kpad, k = idx - n * Kpad;
  float x = (n < N && k < K) ? w[k * N + n] : 0.0f;
  dst[idx] = (_Float16)x;
}

// ---------------------------------------------------------------------------
// Generic WMMA GEMM: C[M x N] = A_f16[M x K] * Bt_f16[Npad x K]^T + bias
// One 16x16 C tile per wave, K in steps of 32 via v_wmma_f32_16x16x32_f16.
// Requires M%16==0, K%32==0, Npad%16==0; writes only cols < N (strided C).
// ---------------------------------------------------------------------------
__global__ void k_gemm(const _Float16* __restrict__ A, const _Float16* __restrict__ Bt,
                       const float* __restrict__ bias, float* __restrict__ C,
                       int M, int N, int Npad, int K, int ldc, int col0, int relu) {
  int wave = (blockIdx.x * blockDim.x + threadIdx.x) >> 5;
  int lane = threadIdx.x & 31;
  int tiles_n = Npad >> 4;
  int tiles = (M >> 4) * tiles_n;
  if (wave >= tiles) return;                     // wave-uniform: EXEC all-1 for WMMA
  int tm = wave / tiles_n, tn = wave - tm * tiles_n;
  int m0 = tm << 4, n0 = tn << 4;
  int half = lane >> 4, l15 = lane & 15;

  const _Float16* arow = A  + (size_t)(m0 + l15) * K + half * 8;   // A frag: K = h*8+j / 16+h*8+j
  const _Float16* brow = Bt + (size_t)(n0 + l15) * K + half * 8;   // B frag: K = h*8+j / 16+h*8+j

  v8f acc = {0.f,0.f,0.f,0.f,0.f,0.f,0.f,0.f};
  for (int k0 = 0; k0 < K; k0 += 32) {
    V16U a, b;
    a.h[0] = *(const v8h*)(arow);
    a.h[1] = *(const v8h*)(arow + 16);
    b.h[0] = *(const v8h*)(brow);
    b.h[1] = *(const v8h*)(brow + 16);
    arow += 32; brow += 32;
    if (k0 + 32 < K) {                            // gfx1250 global_prefetch_b8
      __builtin_prefetch(arow, 0, 1);
      __builtin_prefetch(brow, 0, 1);
    }
    acc = __builtin_amdgcn_wmma_f32_16x16x32_f16(false, a.v, false, b.v,
                                                 (short)0, acc, false, false);
  }
  int col = n0 + l15;
  if (col < N) {
    float bi = bias ? bias[col] : 0.0f;
    for (int r = 0; r < 8; ++r) {
      int row = m0 + half * 8 + r;
      float v = acc[r] + bi;
      if (relu) v = fmaxf(v, 0.0f);
      C[(size_t)row * ldc + col0 + col] = v;
    }
  }
}

// ---------------------------------------------------------------------------
// Softmax attention: one thread per (batch, head, query-row). Two passes over
// keys (online max/denominator, then accumulate). dh <= 64.
// ---------------------------------------------------------------------------
__global__ void k_attn(const float* __restrict__ Q, const float* __restrict__ KV,
                       float* __restrict__ O, int nbt, int lq, int lk, int h, int dh,
                       int q_ld, int kv_ld, int q_off, int k_off, int v_off, float scale) {
  int tid = blockIdx.x * blockDim.x + threadIdx.x;
  int total = nbt * h * lq;
  if (tid >= total) return;
  int qi = tid % lq; int rem = tid / lq; int hh = rem % h; int bt = rem / h;
  const float* q  = Q  + (size_t)(bt * lq + qi) * q_ld + q_off + hh * dh;
  const float* kb = KV + (size_t)bt * lk * kv_ld + k_off + hh * dh;
  const float* vb = KV + (size_t)bt * lk * kv_ld + v_off + hh * dh;

  float m = -1e30f, den = 0.0f;
  for (int kk = 0; kk < lk; ++kk) {
    const float* kp = kb + (size_t)kk * kv_ld;
    float s = 0.0f;
    for (int j = 0; j < dh; ++j) s += q[j] * kp[j];
    s *= scale;
    if (s > m) { den = den * __expf(m - s) + 1.0f; m = s; }
    else       { den += __expf(s - m); }
  }
  float acc[64];
  for (int j = 0; j < dh; ++j) acc[j] = 0.0f;
  for (int kk = 0; kk < lk; ++kk) {
    const float* kp = kb + (size_t)kk * kv_ld;
    const float* vp = vb + (size_t)kk * kv_ld;
    float s = 0.0f;
    for (int j = 0; j < dh; ++j) s += q[j] * kp[j];
    float p = __expf(s * scale - m);
    for (int j = 0; j < dh; ++j) acc[j] += p * vp[j];
  }
  float inv = 1.0f / den;
  float* o = O + (size_t)(bt * lq + qi) * (h * dh) + hh * dh;
  for (int j = 0; j < dh; ++j) o[j] = acc[j] * inv;
}

// out = LayerNorm(a + b) * g + beta    (b may be null; out may alias a). d<=64
__global__ void k_add_ln(const float* __restrict__ a, const float* __restrict__ b,
                         const float* __restrict__ g, const float* __restrict__ bet,
                         float* __restrict__ out, long rows, int d) {
  long r = (long)blockIdx.x * blockDim.x + threadIdx.x;
  if (r >= rows) return;
  float xs[64];
  float mean = 0.0f;
  for (int c = 0; c < d; ++c) {
    float x = a[r * d + c] + (b ? b[r * d + c] : 0.0f);
    xs[c] = x; mean += x;
  }
  mean /= (float)d;
  float var = 0.0f;
  for (int c = 0; c < d; ++c) { float t = xs[c] - mean; var += t * t; }
  var /= (float)d;
  float inv = rsqrtf(var + 1e-5f);
  for (int c = 0; c < d; ++c) out[r * d + c] = (xs[c] - mean) * inv * g[c] + bet[c];
}

// ---------------------------------------------------------------------------
// Small fused feature kernels
// ---------------------------------------------------------------------------
// upper features: cols 8..15 = up2(23->8); all 16 += up3(2->16)
__global__ void k_up_small(const float* __restrict__ upper, const float* __restrict__ w2,
                           const float* __restrict__ b2, const float* __restrict__ w3,
                           const float* __restrict__ b3, float* __restrict__ up) {
  int r = blockIdx.x * blockDim.x + threadIdx.x;
  if (r >= ROWS_UP) return;
  const float* u = upper + (size_t)r * FUP;
  float u0 = u[535], u1 = u[536];
  for (int j = 0; j < 8; ++j) {
    float s = b2[j];
    for (int k = 0; k < 23; ++k) s += u[512 + k] * w2[k * 8 + j];
    up[(size_t)r * 16 + 8 + j] = s;
  }
  for (int c = 0; c < 16; ++c)
    up[(size_t)r * 16 + c] += u0 * w3[c] + u1 * w3[16 + c] + b3[c];
}

// screen features: cols 16..31 = dem2(23->16); all 32 += dem3(2->32) + posenc
__global__ void k_sc_small(const float* __restrict__ scl, const float* __restrict__ w2,
                           const float* __restrict__ b2, const float* __restrict__ w3,
                           const float* __restrict__ b3, float* __restrict__ sc) {
  int r = blockIdx.x * blockDim.x + threadIdx.x;
  if (r >= ROWS_SC) return;
  const float* u = scl + (size_t)r * FSC;
  float u0 = u[537], u1 = u[538];
  for (int j = 0; j < 16; ++j) {
    float s = b2[j];
    for (int k = 0; k < 23; ++k) s += u[512 + k] * w2[k * 16 + j];
    sc[(size_t)r * 32 + 16 + j] = s;
  }
  float pos = (float)(r % NOBJ);
  for (int c = 0; c < 32; ++c) {
    int pair = c >> 1;
    float div = __expf(-(float)(2 * pair) * (9.210340372f / 32.0f));
    float pe = (c & 1) ? __cosf(pos * div) : __sinf(pos * div);
    sc[(size_t)r * 32 + c] += u0 * w3[c] + u1 * w3[32 + c] + b3[c] + pe;
  }
}

// context embeddings: ctx[NBT x 16], ctx_i[NBT x 64]
__global__ void k_ctx(const float* __restrict__ cl,
                      const float* dw, const float* db, const float* hw, const float* hb,
                      const float* aw, const float* ab, const float* iw, const float* ib,
                      float* __restrict__ ctx, float* __restrict__ ctxi) {
  int bt = blockIdx.x * blockDim.x + threadIdx.x;
  if (bt >= NBT) return;
  const float* c = cl + (size_t)bt * NCTX;
  for (int j = 0; j < 4; ++j) {
    float s = db[j];
    for (int k = 0; k < 7; ++k) s += c[k] * dw[k * 4 + j];
    ctx[bt * 16 + j] = s;
  }
  for (int j = 0; j < 4; ++j) {
    float s = hb[j];
    for (int k = 0; k < 24; ++k) s += c[7 + k] * hw[k * 4 + j];
    ctx[bt * 16 + 4 + j] = s;
  }
  for (int j = 0; j < 8; ++j)  ctx[bt * 16 + 8 + j]  = c[32] * aw[j] + ab[j];
  for (int j = 0; j < 64; ++j) ctxi[bt * 64 + j]     = c[31] * iw[j] + ib[j];
}

// seq_input[B,20,64] = concat(up_vec, sc_vec, ctx) + ctx_i[:,T-2]
__global__ void k_seq_input(const float* __restrict__ scl, const int* __restrict__ locl,
                            const float* __restrict__ up_enc, const float* __restrict__ sc_enc,
                            const float* __restrict__ ctx, const float* __restrict__ ctxi,
                            float* __restrict__ si) {
  int r = blockIdx.x * blockDim.x + threadIdx.x;
  if (r >= BB * SEQT) return;
  int b = r / SEQT, t = r % SEQT;
  int bt21 = b * TT + t;
  int loc = locl[bt21];
  int target = (int)scl[((size_t)bt21 * NOBJ + loc) * FSC + (FSC - 1)];
  const float* ci = ctxi + ((size_t)b * TT + (TT - 2)) * 64;
  float* o = si + (size_t)r * 64;
  for (int c = 0; c < 16; ++c) o[c]      = up_enc[((size_t)bt21 * NUP + target) * 16 + c] + ci[c];
  for (int c = 0; c < 32; ++c) o[16 + c] = sc_enc[((size_t)bt21 * NOBJ + loc) * 32 + c]   + ci[16 + c];
  for (int c = 0; c < 16; ++c) o[48 + c] = ctx[(size_t)bt21 * 16 + c]                     + ci[48 + c];
}

// mlp_in[B*NOBJ x 128] = concat(out_next, nxt)
__global__ void k_nxt(const float* __restrict__ scl, const float* __restrict__ up_enc,
                      const float* __restrict__ sc_enc, const float* __restrict__ ctx,
                      const float* __restrict__ ctxi, const float* __restrict__ onext,
                      float* __restrict__ mi) {
  int r = blockIdx.x * blockDim.x + threadIdx.x;
  if (r >= BB * NOBJ) return;
  int b = r / NOBJ, o = r % NOBJ;
  int bt = b * TT + (TT - 1);
  int nt = (int)scl[((size_t)bt * NOBJ + o) * FSC + (FSC - 1)];
  const float* ci = ctxi + (size_t)bt * 64;
  float* d = mi + (size_t)r * 128;
  for (int c = 0; c < 64; ++c) d[c]       = onext[b * 64 + c];
  for (int c = 0; c < 16; ++c) d[64 + c]  = up_enc[((size_t)bt * NUP + nt) * 16 + c] + ci[c];
  for (int c = 0; c < 32; ++c) d[80 + c]  = sc_enc[((size_t)bt * NOBJ + o) * 32 + c] + ci[16 + c];
  for (int c = 0; c < 16; ++c) d[112 + c] = ctx[(size_t)bt * 16 + c]                 + ci[48 + c];
}

// final 128->1 dot per row
__global__ void k_final(const float* __restrict__ hbuf, const float* __restrict__ w,
                        const float* __restrict__ b, float* __restrict__ out) {
  int r = blockIdx.x * blockDim.x + threadIdx.x;
  if (r >= BB * NOBJ) return;
  float s = b[0];
  for (int c = 0; c < 128; ++c) s += hbuf[(size_t)r * 128 + c] * w[c];
  out[r] = s;
}

// ---------------------------------------------------------------------------
// Host-side launch helpers
// ---------------------------------------------------------------------------
static inline int divup(long a, int b) { return (int)((a + b - 1) / b); }

static inline void L_cvt(const float* s, _Float16* d, long rows, int ld, int off,
                         int ks, int kp, hipStream_t st) {
  k_cvt<<<divup(rows * (long)kp, 256), 256, 0, st>>>(s, d, rows, ld, off, ks, kp);
}
static inline void L_wT(const void* w, _Float16* d, int K, int N, int Kp, int Np, hipStream_t st) {
  k_wT<<<divup((long)Np * Kp, 256), 256, 0, st>>>((const float*)w, d, K, N, Kp, Np);
}
static inline void L_gemm(const _Float16* A, const _Float16* Bt, const void* bias, float* C,
                          int M, int N, int Np, int K, int ldc, int col0, int relu, hipStream_t st) {
  int tiles = (M / 16) * (Np / 16);
  k_gemm<<<divup((long)tiles * 32, 128), 128, 0, st>>>(A, Bt, (const float*)bias, C,
                                                      M, N, Np, K, ldc, col0, relu);
}
static inline void L_attn(const float* Q, const float* KV, float* O, int nbt, int lq, int lk,
                          int h, int dh, int qld, int kvld, int qo, int ko, int vo, hipStream_t st) {
  float scale = 1.0f / sqrtf((float)dh);
  k_attn<<<divup((long)nbt * h * lq, 128), 128, 0, st>>>(Q, KV, O, nbt, lq, lk, h, dh,
                                                        qld, kvld, qo, ko, vo, scale);
}
static inline void L_addln(const float* a, const float* b, const void* g, const void* bet,
                           float* out, long rows, int d, hipStream_t st) {
  k_add_ln<<<divup(rows, 128), 128, 0, st>>>(a, b, (const float*)g, (const float*)bet, out, rows, d);
}

// post-norm TransformerEncoderLayer (12 param leaves at din[pb..pb+11])
static void run_enc_layer(float* x, int nbt, int L, int d, int ff, int h,
                          void* const* din, int pb, _Float16* A16, _Float16* W16,
                          float* S0, float* S1, hipStream_t st) {
  int rows = nbt * L;
  int Kd = PAD32(d), Kff = PAD32(ff);
  int N3 = 3 * d, N3p = PAD16(N3), Ndp = PAD16(d), Nfp = PAD16(ff);
  // self-attention
  L_cvt(x, A16, rows, d, 0, d, Kd, st);
  L_wT(din[pb + 0], W16, d, N3, Kd, N3p, st);
  L_gemm(A16, W16, din[pb + 1], S0, rows, N3, N3p, Kd, N3, 0, 0, st);
  L_attn(S0, S0, S1, nbt, L, L, h, d / h, N3, N3, 0, d, 2 * d, st);
  L_cvt(S1, A16, rows, d, 0, d, Kd, st);
  L_wT(din[pb + 2], W16, d, d, Kd, Ndp, st);
  L_gemm(A16, W16, din[pb + 3], S0, rows, d, Ndp, Kd, d, 0, 0, st);
  L_addln(x, S0, din[pb + 4], din[pb + 5], x, rows, d, st);          // ln1
  // feed-forward
  L_cvt(x, A16, rows, d, 0, d, Kd, st);
  L_wT(din[pb + 6], W16, d, ff, Kd, Nfp, st);
  L_gemm(A16, W16, din[pb + 7], S0, rows, ff, Nfp, Kd, ff, 0, 1, st); // relu
  L_cvt(S0, A16, rows, ff, 0, ff, Kff, st);
  L_wT(din[pb + 8], W16, ff, d, Kff, Ndp, st);
  L_gemm(A16, W16, din[pb + 9], S1, rows, d, Ndp, Kff, d, 0, 0, st);
  L_addln(x, S1, din[pb + 10], din[pb + 11], x, rows, d, st);        // ln2
}

// TransformerDecoderLayer (18 param leaves at din[pb..pb+17]); h==1 here
static void run_dec_layer(float* x, const float* mem, int nbt, int Lq, int Lm, int d, int ff,
                          void* const* din, int pb, _Float16* A16, _Float16* W16,
                          float* S0, float* S1, float* S2, hipStream_t st) {
  int rq = nbt * Lq, rm = nbt * Lm;
  int Kd = PAD32(d), Kff = PAD32(ff);
  int N3 = 3 * d, N3p = PAD16(N3), Ndp = PAD16(d), Nfp = PAD16(ff);
  // self-attention on dec
  L_cvt(x, A16, rq, d, 0, d, Kd, st);
  L_wT(din[pb + 0], W16, d, N3, Kd, N3p, st);
  L_gemm(A16, W16, din[pb + 1], S0, rq, N3, N3p, Kd, N3, 0, 0, st);
  L_attn(S0, S0, S1, nbt, Lq, Lq, 1, d, N3, N3, 0, d, 2 * d, st);
  L_cvt(S1, A16, rq, d, 0, d, Kd, st);
  L_wT(din[pb + 2], W16, d, d, Kd, Ndp, st);
  L_gemm(A16, W16, din[pb + 3], S0, rq, d, Ndp, Kd, d, 0, 0, st);
  L_addln(x, S0, din[pb + 4], din[pb + 5], x, rq, d, st);            // ln1
  // cross-attention (q from dec, k/v from mem)
  L_cvt(x, A16, rq, d, 0, d, Kd, st);
  L_wT(din[pb + 6], W16, d, N3, Kd, N3p, st);
  L_gemm(A16, W16, din[pb + 7], S0, rq, N3, N3p, Kd, N3, 0, 0, st);
  L_cvt(mem, A16, rm, d, 0, d, Kd, st);
  L_gemm(A16, W16, din[pb + 7], S1, rm, N3, N3p, Kd, N3, 0, 0, st);  // same W16
  L_attn(S0, S1, S2, nbt, Lq, Lm, 1, d, N3, N3, 0, d, 2 * d, st);
  L_cvt(S2, A16, rq, d, 0, d, Kd, st);
  L_wT(din[pb + 8], W16, d, d, Kd, Ndp, st);
  L_gemm(A16, W16, din[pb + 9], S0, rq, d, Ndp, Kd, d, 0, 0, st);
  L_addln(x, S0, din[pb + 10], din[pb + 11], x, rq, d, st);          // ln2
  // feed-forward
  L_cvt(x, A16, rq, d, 0, d, Kd, st);
  L_wT(din[pb + 12], W16, d, ff, Kd, Nfp, st);
  L_gemm(A16, W16, din[pb + 13], S0, rq, ff, Nfp, Kd, ff, 0, 1, st);
  L_cvt(S0, A16, rq, ff, 0, ff, Kff, st);
  L_wT(din[pb + 14], W16, ff, d, Kff, Ndp, st);
  L_gemm(A16, W16, din[pb + 15], S1, rq, d, Ndp, Kff, d, 0, 0, st);
  L_addln(x, S1, din[pb + 16], din[pb + 17], x, rq, d, st);          // ln3
}

// ---------------------------------------------------------------------------
// kernel_launch
// ---------------------------------------------------------------------------
extern "C" void kernel_launch(void* const* d_in, const int* in_sizes, int n_in,
                              void* d_out, int out_size, void* d_ws, size_t ws_size,
                              hipStream_t stream) {
  (void)in_sizes; (void)n_in; (void)out_size; (void)ws_size;

  // ----- input indices (setup_inputs dict insertion order, params recursive) -----
  const float* scl  = (const float*)d_in[0];
  const int*   locl = (const int*)  d_in[1];
  const float* ctxl = (const float*)d_in[3];
  const float* upl  = (const float*)d_in[4];
  const int P_DEM1 = 5, P_DEM2 = 7, P_DEM3 = 9, P_DAY = 11, P_HOUR = 13;
  const int P_INT = 15, P_APP = 17, P_UP1 = 19, P_UP2 = 21, P_UP3 = 23;
  const int P_TSC = 25, P_TUP = 37;
  const int P_ENCIN = 49, P_DECIN = 51, P_ENC0 = 53, P_ENCNORM = 125;
  const int P_DEC0 = 127, P_DECNORM = 235, P_OUTW = 237;
  const int P_L1 = 239, P_L2 = 241;

  // ----- workspace bump allocation (256B aligned) -----
  char* wsp = (char*)d_ws; size_t off = 0;
  auto alloc = [&](size_t bytes) -> void* {
    void* p = wsp + off;
    off += (bytes + 255) & ~(size_t)255;
    return p;
  };
  _Float16* A16 = (_Float16*)alloc((size_t)ROWS_SC * 512 * 2);   // 68.8 MB
  _Float16* W16 = (_Float16*)alloc(128 * 1024);
  float* XB0   = (float*)alloc((size_t)ROWS_SC * 128 * 4);       // 34.4 MB
  float* XB1   = (float*)alloc((size_t)ROWS_SC * 128 * 4);
  float* UP    = (float*)alloc((size_t)ROWS_UP * 16 * 4);
  float* SC    = (float*)alloc((size_t)ROWS_SC * 32 * 4);
  float* CTX   = (float*)alloc((size_t)NBT * 16 * 4);
  float* CTXI  = (float*)alloc((size_t)NBT * 64 * 4);
  float* SI    = (float*)alloc((size_t)BB * SEQT * 64 * 4);
  float* MEM   = (float*)alloc((size_t)BB * SEQT * 64 * 4);
  float* DEC   = (float*)alloc((size_t)BB * DECL * 64 * 4);
  float* S0    = (float*)alloc((size_t)BB * SEQT * 192 * 4);
  float* S1    = (float*)alloc((size_t)BB * SEQT * 192 * 4);
  float* S2    = (float*)alloc((size_t)BB * SEQT * 64 * 4);
  float* ONEXT = (float*)alloc((size_t)BB * 64 * 4);
  float* MIN_  = (float*)alloc((size_t)BB * NOBJ * 128 * 4);
  float* MH    = (float*)alloc((size_t)BB * NOBJ * 128 * 4);

  // ----- context embeddings -----
  k_ctx<<<divup(NBT, 128), 128, 0, stream>>>(ctxl,
      (const float*)d_in[P_DAY], (const float*)d_in[P_DAY + 1],
      (const float*)d_in[P_HOUR], (const float*)d_in[P_HOUR + 1],
      (const float*)d_in[P_APP], (const float*)d_in[P_APP + 1],
      (const float*)d_in[P_INT], (const float*)d_in[P_INT + 1], CTX, CTXI);

  // ----- upper features: up1 (512->8, WMMA) + up2/up3 (fused) -----
  L_cvt(upl, A16, ROWS_UP, FUP, 0, 512, 512, stream);
  L_wT(d_in[P_UP1], W16, 512, 8, 512, 16, stream);
  L_gemm(A16, W16, d_in[P_UP1 + 1], UP, ROWS_UP, 8, 16, 512, 16, 0, 0, stream);
  k_up_small<<<divup(ROWS_UP, 128), 128, 0, stream>>>(upl,
      (const float*)d_in[P_UP2], (const float*)d_in[P_UP2 + 1],
      (const float*)d_in[P_UP3], (const float*)d_in[P_UP3 + 1], UP);
  // upper encoder (d=16, ff=128, h=2)
  run_enc_layer(UP, NBT, NUP, 16, 128, 2, d_in, P_TUP, A16, W16, XB0, XB1, stream);

  // ----- screen features: dem1 (512->16, WMMA) + dem2/dem3/posenc (fused) -----
  L_cvt(scl, A16, ROWS_SC, FSC, 0, 512, 512, stream);
  L_wT(d_in[P_DEM1], W16, 512, 16, 512, 16, stream);
  L_gemm(A16, W16, d_in[P_DEM1 + 1], SC, ROWS_SC, 16, 16, 512, 32, 0, 0, stream);
  k_sc_small<<<divup(ROWS_SC, 128), 128, 0, stream>>>(scl,
      (const float*)d_in[P_DEM2], (const float*)d_in[P_DEM2 + 1],
      (const float*)d_in[P_DEM3], (const float*)d_in[P_DEM3 + 1], SC);
  // screen encoder (d=32, ff=128, h=2)
  run_enc_layer(SC, NBT, NOBJ, 32, 128, 2, d_in, P_TSC, A16, W16, XB0, XB1, stream);

  // ----- gather + seq_input -----
  k_seq_input<<<divup(BB * SEQT, 128), 128, 0, stream>>>(scl, locl, UP, SC, CTX, CTXI, SI);

  // ----- seq transformer -----
  L_cvt(SI, A16, BB * SEQT, 64, 0, 64, 64, stream);
  L_wT(d_in[P_ENCIN], W16, 64, 64, 64, 64, stream);
  L_gemm(A16, W16, d_in[P_ENCIN + 1], MEM, BB * SEQT, 64, 64, 64, 64, 0, 0, stream);
  k_cvt_dec<<<divup(BB * DECL * 64, 256), 256, 0, stream>>>(SI, A16);
  L_wT(d_in[P_DECIN], W16, 64, 64, 64, 64, stream);
  L_gemm(A16, W16, d_in[P_DECIN + 1], DEC, BB * DECL, 64, 64, 64, 64, 0, 0, stream);

  for (int l = 0; l < 6; ++l)
    run_enc_layer(MEM, BB, SEQT, 64, 128, 1, d_in, P_ENC0 + 12 * l, A16, W16, S0, S1, stream);
  L_addln(MEM, nullptr, d_in[P_ENCNORM], d_in[P_ENCNORM + 1], MEM, BB * SEQT, 64, stream);

  for (int l = 0; l < 6; ++l)
    run_dec_layer(DEC, MEM, BB, DECL, SEQT, 64, 128, d_in, P_DEC0 + 18 * l,
                  A16, W16, S0, S1, S2, stream);
  L_addln(DEC, nullptr, d_in[P_DECNORM], d_in[P_DECNORM + 1], DEC, BB * DECL, 64, stream);

  // out_next = dec.reshape(B,640) @ out_w + b   (M=32, K=640, N=64, WMMA)
  L_cvt(DEC, A16, BB, DECL * 64, 0, DECL * 64, DECL * 64, stream);
  L_wT(d_in[P_OUTW], W16, DECL * 64, 64, DECL * 64, 64, stream);
  L_gemm(A16, W16, d_in[P_OUTW + 1], ONEXT, BB, 64, 64, DECL * 64, 64, 0, 0, stream);

  // ----- next-screen branch + MLP -----
  k_nxt<<<divup(BB * NOBJ, 128), 128, 0, stream>>>(scl, UP, SC, CTX, CTXI, ONEXT, MIN_);
  L_cvt(MIN_, A16, BB * NOBJ, 128, 0, 128, 128, stream);
  L_wT(d_in[P_L1], W16, 128, 128, 128, 128, stream);
  L_gemm(A16, W16, d_in[P_L1 + 1], MH, BB * NOBJ, 128, 128, 128, 128, 0, 1, stream); // relu
  k_final<<<divup(BB * NOBJ, 128), 128, 0, stream>>>(MH,
      (const float*)d_in[P_L2], (const float*)d_in[P_L2 + 1], (float*)d_out);
}
